// TritonGroupedExpertsFused_18451179504157
// MI455X (gfx1250) — compile-verified, matrix operand order checked by reading, not run on hardware
//
#include <hip/hip_runtime.h>
#include <hip/hip_bf16.h>

typedef __attribute__((ext_vector_type(16))) __bf16 v16bf;
typedef __attribute__((ext_vector_type(8)))  float  v8f;
typedef __attribute__((vector_size(16)))     int    vsi4;   // async-LDS builtin operand type

#define N_TOKENS  8192
#define D_MODEL   1024
#define D_FF      4096
#define N_EXPERTS 8
#define TOP_K     2
#define PAIRS     (N_TOKENS * TOP_K)          // 16384 routed (token, slot) pairs
#define RPAD      (PAIRS + N_EXPERTS * 64)    // 16896: worst-case padded rows
#define BM        64
#define BN        64
#define LDT       40                          // LDS row stride in halves (80B: 16B-aligned, bank-spread)
#define MBLK      (RPAD / BM)                 // 264 fixed M-blocks

// ---- workspace layout (bytes, 256B aligned) ----
#define WS_OFFS     0                                    // int offs[9]
#define WS_PAIRROW  256                                  // int pair_row[PAIRS]
#define WS_LISTTOK  (WS_PAIRROW + PAIRS * 4)             // int list_token[RPAD]
#define WS_H        (WS_LISTTOK + RPAD * 4)              // = 133376 (256-mult): bf16 h[RPAD][D_FF]
#define WS_Y        (WS_H + (size_t)RPAD * D_FF * 2)     // f32 y[RPAD][D_MODEL]

#if __has_builtin(__builtin_amdgcn_global_load_async_to_lds_b128) && \
    __has_builtin(__builtin_amdgcn_s_wait_asynccnt)
#define ASYNC_LDS 1
#endif

// -------------------------------------------------------------------------
// helpers
// -------------------------------------------------------------------------
// Native f32 -> bf16 (RNE) pack of two values into one dword.
__device__ __forceinline__ unsigned pkbf2(float lo, float hi) {
  union { __bf16 h[2]; unsigned u; } p;
  p.h[0] = (__bf16)lo;
  p.h[1] = (__bf16)hi;
  return p.u;
}

union FragU { v16bf bf; uint4 u4[2]; };

// Load one 16-half WMMA fragment from LDS.
// Packing (mirrors ISA 16-bit A-matrix table): lanes 0-15 hold K=0..7,16..23;
// lanes 16-31 hold K=8..15,24..31 -> two contiguous 16B chunks per lane.
__device__ __forceinline__ v16bf ld_frag(const __bf16* lds, int row, int lane) {
  const __bf16* p = lds + row * LDT + ((lane & 16) ? 8 : 0);
  FragU f;
  f.u4[0] = *(const uint4*)(p);
  f.u4[1] = *(const uint4*)(p + 16);
  return f.bf;
}

__device__ __forceinline__ v8f wmma_bf16(v16bf a, v16bf b, v8f c) {
  return __builtin_amdgcn_wmma_f32_16x16x32_bf16(false, a, false, b, (short)0, c, false, false);
}

// -------------------------------------------------------------------------
// 1) deterministic routing: per-expert 64-aligned segments
// -------------------------------------------------------------------------
__global__ void __launch_bounds__(32)
route_kernel(const int* __restrict__ eidx, int* __restrict__ offs,
             int* __restrict__ pair_row, int* __restrict__ tok) {
  __shared__ int cntS[N_EXPERTS];
  __shared__ int offsS[N_EXPERTS + 1];
  __shared__ int fillS[N_EXPERTS];
  const int lane = threadIdx.x;
  if (lane < N_EXPERTS) { cntS[lane] = 0; fillS[lane] = 0; }
  __syncthreads();

  for (int base = 0; base < PAIRS; base += 32) {
    int e = eidx[base + lane];
    atomicAdd(&cntS[e], 1);                     // integer add: order-independent
  }
  __syncthreads();

  if (lane == 0) {
    int o = 0;
    for (int x = 0; x < N_EXPERTS; x++) { offsS[x] = o; o += (cntS[x] + 63) & ~63; }
    offsS[N_EXPERTS] = o;
    for (int x = 0; x <= N_EXPERTS; x++) offs[x] = offsS[x];
  }
  __syncthreads();

  // stable (pair-order) rank assignment via wave32 ballots -> deterministic rows
  for (int base = 0; base < PAIRS; base += 32) {
    int e = eidx[base + lane];
    int row = 0;
    for (int x = 0; x < N_EXPERTS; x++) {
      unsigned long long m = __ballot(e == x);
      if (e == x)
        row = offsS[x] + fillS[x] + __popcll(m & ((1ull << lane) - 1ull));
      __syncthreads();
      if (lane == 0) fillS[x] += __popcll(m);
      __syncthreads();
    }
    pair_row[base + lane] = row;
    tok[row] = (base + lane) >> 1;              // token index = pair / TOP_K
  }
  __syncthreads();

  // mark padded rows
  for (int x = 0; x < N_EXPERTS; x++)
    for (int r = offsS[x] + cntS[x] + lane; r < offsS[x + 1]; r += 32)
      tok[r] = -1;
}

// -------------------------------------------------------------------------
// 2) h = silu(xg @ w1[e]) * (xg @ w2[e])   (grouped GEMM, bf16 WMMA, f32 acc)
// -------------------------------------------------------------------------
__global__ void __launch_bounds__(256)
ffn1_kernel(const float* __restrict__ x, const float* __restrict__ w1,
            const float* __restrict__ w2, const int* __restrict__ offs,
            const int* __restrict__ tok, __bf16* __restrict__ hbuf) {
  __shared__ __align__(16) __bf16 sA [BM * LDT];
  __shared__ __align__(16) __bf16 sB1[BN * LDT];
  __shared__ __align__(16) __bf16 sB2[BN * LDT];

  const int row0 = blockIdx.x * BM;
  if (row0 >= offs[N_EXPERTS]) return;
  int e = 0;
  while (row0 >= offs[e + 1]) e++;

  const int n0 = blockIdx.y * BN;
  const float* w1e = w1 + (size_t)e * D_MODEL * D_FF;
  const float* w2e = w2 + (size_t)e * D_MODEL * D_FF;

  const int tid  = threadIdx.x;
  const int lane = tid & 31;
  const int wm   = (tid >> 5) & 3;      // M sub-tile (4 x 16 rows)
  const int wn   = tid >> 7;            // N slab (2 x 32 cols)

  const int ra = tid >> 2;              // A-stage row 0..63
  const int ka = (tid & 3) << 3;        // A-stage k offset 0/8/16/24
  const int myTok = tok[row0 + ra];

  const int kb = tid >> 3;              // B-stage k 0..31
  const int nb = (tid & 7) << 3;        // B-stage n offset 0..56

  v8f g0 = {0,0,0,0,0,0,0,0}, g1 = g0, v0 = g0, v1 = g0;

  for (int kk = 0; kk < D_MODEL; kk += 32) {
    // stage A: gather token row, f32 -> bf16 (zeros for padded rows)
    uint4 pk;
    if (myTok >= 0) {
      const float4 f0 = *(const float4*)(x + (size_t)myTok * D_MODEL + kk + ka);
      const float4 f1 = *(const float4*)(x + (size_t)myTok * D_MODEL + kk + ka + 4);
      pk.x = pkbf2(f0.x, f0.y);
      pk.y = pkbf2(f0.z, f0.w);
      pk.z = pkbf2(f1.x, f1.y);
      pk.w = pkbf2(f1.z, f1.w);
    } else {
      pk.x = 0u; pk.y = 0u; pk.z = 0u; pk.w = 0u;
    }
    *(uint4*)(&sA[ra * LDT + ka]) = pk;

    // stage B1/B2: f32 weights -> bf16, transposed to [n][k]
    const float* p1 = w1e + (size_t)(kk + kb) * D_FF + n0 + nb;
    const float* p2 = w2e + (size_t)(kk + kb) * D_FF + n0 + nb;
    {
      const float4 a0 = *(const float4*)(p1);
      const float4 a1 = *(const float4*)(p1 + 4);
      sB1[(nb + 0) * LDT + kb] = (__bf16)a0.x;
      sB1[(nb + 1) * LDT + kb] = (__bf16)a0.y;
      sB1[(nb + 2) * LDT + kb] = (__bf16)a0.z;
      sB1[(nb + 3) * LDT + kb] = (__bf16)a0.w;
      sB1[(nb + 4) * LDT + kb] = (__bf16)a1.x;
      sB1[(nb + 5) * LDT + kb] = (__bf16)a1.y;
      sB1[(nb + 6) * LDT + kb] = (__bf16)a1.z;
      sB1[(nb + 7) * LDT + kb] = (__bf16)a1.w;
      const float4 b0 = *(const float4*)(p2);
      const float4 b1 = *(const float4*)(p2 + 4);
      sB2[(nb + 0) * LDT + kb] = (__bf16)b0.x;
      sB2[(nb + 1) * LDT + kb] = (__bf16)b0.y;
      sB2[(nb + 2) * LDT + kb] = (__bf16)b0.z;
      sB2[(nb + 3) * LDT + kb] = (__bf16)b0.w;
      sB2[(nb + 4) * LDT + kb] = (__bf16)b1.x;
      sB2[(nb + 5) * LDT + kb] = (__bf16)b1.y;
      sB2[(nb + 6) * LDT + kb] = (__bf16)b1.z;
      sB2[(nb + 7) * LDT + kb] = (__bf16)b1.w;
    }
    if (kk + 32 < D_MODEL) {
      __builtin_prefetch(p1 + (size_t)32 * D_FF, 0, 0);   // global_prefetch next K slab
      __builtin_prefetch(p2 + (size_t)32 * D_FF, 0, 0);
    }
    __syncthreads();

    const v16bf a   = ld_frag(sA,  wm * 16 +      (lane & 15), lane);
    const v16bf b10 = ld_frag(sB1, wn * 32 +      (lane & 15), lane);
    const v16bf b11 = ld_frag(sB1, wn * 32 + 16 + (lane & 15), lane);
    const v16bf b20 = ld_frag(sB2, wn * 32 +      (lane & 15), lane);
    const v16bf b21 = ld_frag(sB2, wn * 32 + 16 + (lane & 15), lane);
    g0 = wmma_bf16(a, b10, g0);
    g1 = wmma_bf16(a, b11, g1);
    v0 = wmma_bf16(a, b20, v0);
    v1 = wmma_bf16(a, b21, v1);
    __syncthreads();
  }

  // epilogue: h = silu(g) * v -> bf16 (C layout: M = r + 8*(lane>=16), N = lane%16)
  const int mBase = row0 + wm * 16 + ((lane >> 4) << 3);
  const int cBase = n0 + wn * 32 + (lane & 15);
  for (int r = 0; r < 8; r++) {
    const size_t gr = (size_t)(mBase + r) * D_FF;
    float g = g0[r];
    hbuf[gr + cBase]      = (__bf16)(g * (1.0f / (1.0f + __expf(-g))) * v0[r]);
    g = g1[r];
    hbuf[gr + cBase + 16] = (__bf16)(g * (1.0f / (1.0f + __expf(-g))) * v1[r]);
  }
}

// -------------------------------------------------------------------------
// 3) y = h @ w3[e]
// -------------------------------------------------------------------------
__global__ void __launch_bounds__(256)
ffn2_kernel(const __bf16* __restrict__ hbuf, const float* __restrict__ w3,
            const int* __restrict__ offs, float* __restrict__ ybuf) {
  __shared__ __align__(16) __bf16 sA[BM * LDT];
  __shared__ __align__(16) __bf16 sB[BN * LDT];

  const int row0 = blockIdx.x * BM;
  if (row0 >= offs[N_EXPERTS]) return;
  int e = 0;
  while (row0 >= offs[e + 1]) e++;

  const int n0 = blockIdx.y * BN;
  const float* w3e = w3 + (size_t)e * D_FF * D_MODEL;

  const int tid  = threadIdx.x;
  const int lane = tid & 31;
  const int wm   = (tid >> 5) & 3;
  const int wn   = tid >> 7;

  const int ra = tid >> 2;
  const int ka = (tid & 3) << 3;
  const int kb = tid >> 3;
  const int nb = (tid & 7) << 3;

  v8f y0 = {0,0,0,0,0,0,0,0}, y1 = y0;

  for (int kk = 0; kk < D_FF; kk += 32) {
    // stage A: h already bf16 -> pure 16B copy; use CDNA5 async global->LDS path
#if defined(ASYNC_LDS)
    __builtin_amdgcn_global_load_async_to_lds_b128(
        (__attribute__((address_space(1))) vsi4*)(hbuf + (size_t)(row0 + ra) * D_FF + kk + ka),
        (__attribute__((address_space(3))) vsi4*)(&sA[ra * LDT + ka]),
        0, 0);
#else
    *(uint4*)(&sA[ra * LDT + ka]) =
        *(const uint4*)(hbuf + (size_t)(row0 + ra) * D_FF + kk + ka);
#endif

    // stage B: w3 f32 -> bf16 transposed
    const float* p = w3e + (size_t)(kk + kb) * D_MODEL + n0 + nb;
    {
      const float4 a0 = *(const float4*)(p);
      const float4 a1 = *(const float4*)(p + 4);
      sB[(nb + 0) * LDT + kb] = (__bf16)a0.x;
      sB[(nb + 1) * LDT + kb] = (__bf16)a0.y;
      sB[(nb + 2) * LDT + kb] = (__bf16)a0.z;
      sB[(nb + 3) * LDT + kb] = (__bf16)a0.w;
      sB[(nb + 4) * LDT + kb] = (__bf16)a1.x;
      sB[(nb + 5) * LDT + kb] = (__bf16)a1.y;
      sB[(nb + 6) * LDT + kb] = (__bf16)a1.z;
      sB[(nb + 7) * LDT + kb] = (__bf16)a1.w;
    }
    if (kk + 32 < D_FF)
      __builtin_prefetch(p + (size_t)32 * D_MODEL, 0, 0);
#if defined(ASYNC_LDS)
    __builtin_amdgcn_s_wait_asynccnt(0);
#endif
    __syncthreads();

    const v16bf a  = ld_frag(sA, wm * 16 +      (lane & 15), lane);
    const v16bf b0 = ld_frag(sB, wn * 32 +      (lane & 15), lane);
    const v16bf b1 = ld_frag(sB, wn * 32 + 16 + (lane & 15), lane);
    y0 = wmma_bf16(a, b0, y0);
    y1 = wmma_bf16(a, b1, y1);
    __syncthreads();
  }

  const int mBase = row0 + wm * 16 + ((lane >> 4) << 3);
  const int cBase = n0 + wn * 32 + (lane & 15);
  for (int r = 0; r < 8; r++) {
    const size_t gr = (size_t)(mBase + r) * D_MODEL;
    ybuf[gr + cBase]      = y0[r];
    ybuf[gr + cBase + 16] = y1[r];
  }
}

// -------------------------------------------------------------------------
// 4) out[t] = p0 * y[row(t,0)] + p1 * y[row(t,1)]   (fixed order -> deterministic)
// -------------------------------------------------------------------------
__global__ void __launch_bounds__(256)
combine_kernel(const float* __restrict__ ybuf, const int* __restrict__ pair_row,
               const float* __restrict__ ew, float* __restrict__ out) {
  const int t = blockIdx.x;
  const int d = threadIdx.x << 2;               // 256 threads x 4 floats = 1024
  const int r0 = pair_row[t * 2 + 0];
  const int r1 = pair_row[t * 2 + 1];
  const float p0 = ew[t * 2 + 0];
  const float p1 = ew[t * 2 + 1];
  const float4 a = *(const float4*)(ybuf + (size_t)r0 * D_MODEL + d);
  const float4 b = *(const float4*)(ybuf + (size_t)r1 * D_MODEL + d);
  float4 o;
  o.x = p0 * a.x + p1 * b.x;
  o.y = p0 * a.y + p1 * b.y;
  o.z = p0 * a.z + p1 * b.z;
  o.w = p0 * a.w + p1 * b.w;
  *(float4*)(out + (size_t)t * D_MODEL + d) = o;
}

// -------------------------------------------------------------------------
extern "C" void kernel_launch(void* const* d_in, const int* in_sizes, int n_in,
                              void* d_out, int out_size, void* d_ws, size_t ws_size,
                              hipStream_t stream) {
  const float* x    = (const float*)d_in[0];
  const int*   eidx = (const int*)  d_in[1];
  const float* ew   = (const float*)d_in[2];
  const float* w1   = (const float*)d_in[3];
  const float* w2   = (const float*)d_in[4];
  const float* w3   = (const float*)d_in[5];
  float* out = (float*)d_out;

  char* ws = (char*)d_ws;
  int*    offs = (int*)(ws + WS_OFFS);
  int*    prow = (int*)(ws + WS_PAIRROW);
  int*    ltok = (int*)(ws + WS_LISTTOK);
  __bf16* hbuf = (__bf16*)(ws + WS_H);
  float*  ybuf = (float*)(ws + WS_Y);

  route_kernel<<<1, 32, 0, stream>>>(eidx, offs, prow, ltok);
  ffn1_kernel<<<dim3(MBLK, D_FF / BN), 256, 0, stream>>>(x, w1, w2, offs, ltok, hbuf);
  ffn2_kernel<<<dim3(MBLK, D_MODEL / BN), 256, 0, stream>>>(hbuf, w3, offs, ybuf);
  combine_kernel<<<N_TOKENS, 256, 0, stream>>>(ybuf, prow, ew, out);
}
